// ImageMultiheadCrossAttention_69303592288486
// MI455X (gfx1250) — matrix operand order repaired and checked by reading, not compiled, not run on hardware
//
#include <hip/hip_runtime.h>

// ---------------------------------------------------------------------------
// CDNA5 (gfx1250, wave32) flash-attention implementation of the reference:
//   Q/K/V = 1x1 conv projections (64x512 GEMM), energy = Q^T K (4096x4096),
//   softmax, out = attn * V^T, broadcast across 8 heads.
// Matmuls via v_wmma_f32_16x16x32_f16; K/V tile staging via
// global_load_async_to_lds_b128 (ASYNCcnt) when available.
// ---------------------------------------------------------------------------

typedef __attribute__((ext_vector_type(16))) _Float16 v16h;
typedef __attribute__((ext_vector_type(4)))  _Float16 v4h;
typedef __attribute__((ext_vector_type(8)))  float    v8f;
typedef __attribute__((ext_vector_type(4)))  int      i4;

typedef __attribute__((address_space(1))) i4* gp_i4;  // global (non-const)
typedef __attribute__((address_space(3))) i4* lp_i4;  // LDS

#if __has_builtin(__builtin_amdgcn_global_load_async_to_lds_b128)
#define HAVE_ASYNC_LDS 1
#else
#define HAVE_ASYNC_LDS 0
#endif

union HReg { v16h v; v4h q[4]; _Float16 h[16]; };

__device__ __forceinline__ v8f wmma_f16(const HReg& a, const HReg& b, v8f c) {
  // 8 args: (neg_a, A, neg_b, B, c_mod, C, reuse_a, reuse_b)
  return __builtin_amdgcn_wmma_f32_16x16x32_f16(false, a.v, false, b.v,
                                                (short)0, c, false, false);
}

__device__ __forceinline__ v8f zero8() {
  v8f z;
#pragma unroll
  for (int i = 0; i < 8; ++i) z[i] = 0.f;
  return z;
}

// ---------------------------------------------------------------------------
// Phase 1: projections. One workgroup = (batch b, proj p, 128-wide hw block).
// Computes Y = W(64x512) * X(512x4096) + bias for its block, stores f16:
//   p=0 -> Qws[b][n][d]   (row-major over d; A-operand friendly)
//   p=1 -> Kws[b][m][d]
//   p=2 -> Vt [b][d][m]
// ---------------------------------------------------------------------------
__global__ __launch_bounds__(256) void proj_qkv_kernel(
    const float* __restrict__ qx, const float* __restrict__ kx,
    const float* __restrict__ vx,
    const float* __restrict__ wq, const float* __restrict__ wk,
    const float* __restrict__ wv,
    const float* __restrict__ bq, const float* __restrict__ bk,
    const float* __restrict__ bv,
    _Float16* __restrict__ Qws, _Float16* __restrict__ Kws,
    _Float16* __restrict__ Vt)
{
  constexpr int C = 512, HW = 4096, XP = 40;  // XP: padded LDS row (halves)
  __shared__ _Float16 Wl[64 * 32];     // W chunk  [d][c]      4 KB
  __shared__ _Float16 Xl[128 * XP];    // X chunk  [hw][c] pad 10 KB

  const int tid  = threadIdx.x;
  const int wave = tid >> 5;
  const int lane = tid & 31;
  const int l16  = lane & 15;
  const int hi   = lane >> 4;
  const int h8   = hi * 8;
  const int h16  = hi * 16;

  const int p   = blockIdx.y;
  const int b   = blockIdx.z;
  const int hwb = blockIdx.x * 128;

  const float* __restrict__ x  = (p == 0) ? qx : (p == 1) ? kx : vx;
  const float* __restrict__ w  = (p == 0) ? wq : (p == 1) ? wk : wv;
  const float* __restrict__ bs = (p == 0) ? bq : (p == 1) ? bk : bv;

  v8f acc[4];
#pragma unroll
  for (int dt = 0; dt < 4; ++dt) acc[dt] = zero8();

  const size_t xbase = (size_t)b * C * HW + hwb;

  for (int c0 = 0; c0 < C; c0 += 32) {
    // Stage W chunk [64 d][32 c], f32 -> f16 (coalesced along c)
#pragma unroll
    for (int it = 0; it < 8; ++it) {
      int i = tid + 256 * it;
      int d = i >> 5, c = i & 31;
      Wl[i] = (_Float16)w[d * C + c0 + c];
    }
    // Stage X chunk transposed: Xl[hw][c] (global coalesced along hw)
#pragma unroll
    for (int it = 0; it < 16; ++it) {
      int i  = tid + 256 * it;
      int c  = i >> 7, hw = i & 127;
      Xl[hw * XP + c] = (_Float16)x[xbase + (size_t)(c0 + c) * HW + hw];
    }
    // Prefetch next X chunk toward cache while this chunk is consumed
    if (c0 + 32 < C) {
      int pc  = tid >> 3;               // 0..31
      int phw = (tid & 7) << 4;         // 0..112
      __builtin_prefetch(&x[xbase + (size_t)(c0 + 32 + pc) * HW + phw], 0, 1);
    }
    __syncthreads();

    // B operand: X, N = hw column (lane&15), K = c = e + 16*hi
    HReg bx;
    {
      const _Float16* xr = Xl + (wave * 16 + l16) * XP + h16;
      bx.q[0] = *(const v4h*)(xr + 0);
      bx.q[1] = *(const v4h*)(xr + 4);
      bx.q[2] = *(const v4h*)(xr + 8);
      bx.q[3] = *(const v4h*)(xr + 12);
    }
    // A operand: W, M = d row (lane&15), K = c = e + 16*hi
#pragma unroll
    for (int dt = 0; dt < 4; ++dt) {
      const _Float16* wr = Wl + (16 * dt + l16) * 32;
      HReg aw;
      aw.q[0] = *(const v4h*)(wr + h16);
      aw.q[1] = *(const v4h*)(wr + h16 + 4);
      aw.q[2] = *(const v4h*)(wr + h16 + 8);
      aw.q[3] = *(const v4h*)(wr + h16 + 12);
      acc[dt] = wmma_f16(aw, bx, acc[dt]);
    }
    __syncthreads();
  }

  // Epilogue: add bias, store f16. C/D layout: col = lane&15, row = r + 8*hi.
  const int hw = hwb + wave * 16 + l16;
  if (p < 2) {
    _Float16* dst = ((p == 0) ? Qws : Kws) + ((size_t)b * HW + hw) * 64;
#pragma unroll
    for (int dt = 0; dt < 4; ++dt)
#pragma unroll
      for (int r = 0; r < 8; ++r) {
        int d = 16 * dt + r + h8;
        dst[d] = (_Float16)(acc[dt][r] + bs[d]);
      }
  } else {
#pragma unroll
    for (int dt = 0; dt < 4; ++dt)
#pragma unroll
      for (int r = 0; r < 8; ++r) {
        int d = 16 * dt + r + h8;
        Vt[((size_t)b * 64 + d) * HW + hw] = (_Float16)(acc[dt][r] + bs[d]);
      }
  }
}

// ---------------------------------------------------------------------------
// Phase 2: flash attention. One workgroup = (batch b, 128-query block).
// 8 waves; each wave owns 16 query rows. Loop over 32 key tiles of 128.
// Never materializes the 4096x4096 energy matrix.
// ---------------------------------------------------------------------------
__global__ __launch_bounds__(256) void attn_kernel(
    const _Float16* __restrict__ Qws, const _Float16* __restrict__ Kws,
    const _Float16* __restrict__ Vt, float* __restrict__ out)
{
  constexpr int HW = 4096, D = 64, BM = 128, BN = 128;
  __shared__ __align__(16) float smem[16384];      // 64 KB total
  _Float16* ldsK = (_Float16*)smem;                // [BN][D]   16 KB
  _Float16* ldsV = (_Float16*)smem + BN * D;       // [D][BN]   16 KB
  _Float16* ldsP = (_Float16*)smem + 2 * BN * D;   // [wave][16][BN] 32 KB
  float*    ldsO = smem;                           // [D][BM] reuse of K+V

  const int tid  = threadIdx.x;
  const int wave = tid >> 5;
  const int lane = tid & 31;
  const int l16  = lane & 15;
  const int hi   = lane >> 4;
  const int h8   = hi * 8;
  const int h16  = hi * 16;

  const int b  = blockIdx.y;
  const int nb = blockIdx.x * BM;

  // Q A-operand regs (reused for all 32 key tiles): M = query row, K = d (2x32)
  HReg aq[2];
  {
    const _Float16* qrow = Qws + ((size_t)b * HW + nb + wave * 16 + l16) * D;
#pragma unroll
    for (int t = 0; t < 2; ++t) {
      aq[t].q[0] = *(const v4h*)(qrow + 32 * t + h16);
      aq[t].q[1] = *(const v4h*)(qrow + 32 * t + h16 + 4);
      aq[t].q[2] = *(const v4h*)(qrow + 32 * t + h16 + 8);
      aq[t].q[3] = *(const v4h*)(qrow + 32 * t + h16 + 12);
    }
  }

  float mrow[8], lrow[8];
  v8f oacc[4];
#pragma unroll
  for (int r = 0; r < 8; ++r) { mrow[r] = -__builtin_inff(); lrow[r] = 0.f; }
#pragma unroll
  for (int dt = 0; dt < 4; ++dt) oacc[dt] = zero8();

  _Float16* pbuf = ldsP + wave * (16 * BN);

  for (int mb = 0; mb < HW; mb += BN) {
    // Cooperative stage: K tile [m][d], V tile [d][m]
    {
      const _Float16* ksrc = Kws + ((size_t)b * HW + mb) * D;
      const _Float16* vsrc = Vt + (size_t)b * 64 * HW + mb;
#if HAVE_ASYNC_LDS
      // Async global->LDS B128 copies, tracked by ASYNCcnt (no VGPR data path)
#pragma unroll
      for (int it = 0; it < 4; ++it) {
        int i = tid + 256 * it;              // 1024 x 16B = 16 KB (K tile)
        int m = i >> 3, c8 = (i & 7) << 3;
        __builtin_amdgcn_global_load_async_to_lds_b128(
            (gp_i4)(ksrc + m * D + c8),
            (lp_i4)(void*)(ldsK + m * D + c8), 0, 0);
      }
#pragma unroll
      for (int it = 0; it < 4; ++it) {
        int i = tid + 256 * it;              // 1024 x 16B = 16 KB (V tile)
        int d = i >> 4, m8 = (i & 15) << 3;
        __builtin_amdgcn_global_load_async_to_lds_b128(
            (gp_i4)(vsrc + (size_t)d * HW + m8),
            (lp_i4)(void*)(ldsV + d * BN + m8), 0, 0);
      }
      asm volatile("s_wait_asynccnt 0x0" ::: "memory");
#else
#pragma unroll
      for (int it = 0; it < 8; ++it) {
        int i = tid + 256 * it;              // 2048 v4h = BN*D halves
        int m = i >> 4, c4 = (i & 15) << 2;
        *(v4h*)(ldsK + m * D + c4) = *(const v4h*)(ksrc + m * D + c4);
      }
#pragma unroll
      for (int it = 0; it < 8; ++it) {
        int i = tid + 256 * it;
        int d = i >> 5, m4 = (i & 31) << 2;
        *(v4h*)(ldsV + d * BN + m4) = *(const v4h*)(vsrc + (size_t)d * HW + m4);
      }
#endif
    }
    __syncthreads();

    // S = Q * K^T : 8 tiles of 16x16, K-dim = 64 (2 WMMAs each)
    v8f S[8];
#pragma unroll
    for (int j = 0; j < 8; ++j) {
      const _Float16* krow = ldsK + (16 * j + l16) * D;  // N = m column
      v8f c = zero8();
#pragma unroll
      for (int t = 0; t < 2; ++t) {
        HReg bk;
        const _Float16* kp = krow + 32 * t + h16;        // K = d = 32t+h16+e
        bk.q[0] = *(const v4h*)(kp + 0);
        bk.q[1] = *(const v4h*)(kp + 4);
        bk.q[2] = *(const v4h*)(kp + 8);
        bk.q[3] = *(const v4h*)(kp + 12);
        c = wmma_f16(aq[t], bk, c);
      }
      S[j] = c;
    }

    // Online softmax. C/D layout: lane holds rows r+8*hi, column = lane&15.
    float rmax[8];
#pragma unroll
    for (int r = 0; r < 8; ++r) {
      float m = S[0][r];
#pragma unroll
      for (int j = 1; j < 8; ++j) m = fmaxf(m, S[j][r]);
      rmax[r] = m;
    }
#pragma unroll
    for (int off = 1; off < 16; off <<= 1)
#pragma unroll
      for (int r = 0; r < 8; ++r)
        rmax[r] = fmaxf(rmax[r], __shfl_xor(rmax[r], off, 32));

    float scale[8], psum[8];
#pragma unroll
    for (int r = 0; r < 8; ++r) {
      float mnew = fmaxf(mrow[r], rmax[r]);
      scale[r] = __expf(mrow[r] - mnew);
      mrow[r]  = mnew;
      psum[r]  = 0.f;
    }
#pragma unroll
    for (int j = 0; j < 8; ++j)
#pragma unroll
      for (int r = 0; r < 8; ++r) {
        float pe = __expf(S[j][r] - mrow[r]);
        S[j][r] = pe;
        psum[r] += pe;
      }
#pragma unroll
    for (int off = 1; off < 16; off <<= 1)
#pragma unroll
      for (int r = 0; r < 8; ++r)
        psum[r] += __shfl_xor(psum[r], off, 32);
#pragma unroll
    for (int r = 0; r < 8; ++r) lrow[r] = lrow[r] * scale[r] + psum[r];
#pragma unroll
    for (int dt = 0; dt < 4; ++dt)
#pragma unroll
      for (int r = 0; r < 8; ++r) oacc[dt][r] *= scale[r];

    // Re-layout P: C/D -> A via per-wave LDS scratch (f16, [16][BN])
#pragma unroll
    for (int j = 0; j < 8; ++j)
#pragma unroll
      for (int r = 0; r < 8; ++r)
        pbuf[(r + h8) * BN + 16 * j + l16] = (_Float16)S[j][r];
    asm volatile("s_wait_dscnt 0" ::: "memory");

    HReg pa[4];
    {
      const _Float16* prow = pbuf + l16 * BN;            // M = query row
#pragma unroll
      for (int jj = 0; jj < 4; ++jj) {                   // K = m, 4x32
        pa[jj].q[0] = *(const v4h*)(prow + 32 * jj + h16);
        pa[jj].q[1] = *(const v4h*)(prow + 32 * jj + h16 + 4);
        pa[jj].q[2] = *(const v4h*)(prow + 32 * jj + h16 + 8);
        pa[jj].q[3] = *(const v4h*)(prow + 32 * jj + h16 + 12);
      }
    }

    // O += P * V^T : 4 d-tiles, K-dim = 128 (4 WMMAs each)
#pragma unroll
    for (int dt = 0; dt < 4; ++dt) {
      const _Float16* vrow = ldsV + (16 * dt + l16) * BN;  // N = d column
#pragma unroll
      for (int jj = 0; jj < 4; ++jj) {
        HReg bv;
        const _Float16* vp = vrow + 32 * jj + h16;         // K = m
        bv.q[0] = *(const v4h*)(vp + 0);
        bv.q[1] = *(const v4h*)(vp + 4);
        bv.q[2] = *(const v4h*)(vp + 8);
        bv.q[3] = *(const v4h*)(vp + 12);
        oacc[dt] = wmma_f16(pa[jj], bv, oacc[dt]);
      }
    }
    __syncthreads();
  }

  // Finalize: O /= l, transpose through LDS, broadcast-store 8 heads.
#pragma unroll
  for (int r = 0; r < 8; ++r) lrow[r] = 1.f / lrow[r];
#pragma unroll
  for (int dt = 0; dt < 4; ++dt)
#pragma unroll
    for (int r = 0; r < 8; ++r) {
      int n = wave * 16 + r + h8;
      int d = 16 * dt + l16;
      ldsO[d * BM + n] = oacc[dt][r] * lrow[r];
    }
  __syncthreads();

  float* outb = out + (size_t)b * 512 * HW + nb;
#pragma unroll
  for (int it = 0; it < 8; ++it) {
    int lin = tid + 256 * it;          // 2048 float4 per head
    int d   = lin >> 5;
    int n4  = (lin & 31) << 2;
    float4 val = *(const float4*)(ldsO + d * BM + n4);
#pragma unroll
    for (int h = 0; h < 8; ++h)
      *(float4*)(outb + (size_t)(h * 64 + d) * HW + n4) = val;
  }
}

// ---------------------------------------------------------------------------
extern "C" void kernel_launch(void* const* d_in, const int* in_sizes, int n_in,
                              void* d_out, int out_size, void* d_ws,
                              size_t ws_size, hipStream_t stream) {
  (void)in_sizes; (void)n_in; (void)out_size; (void)ws_size;
  const float* q  = (const float*)d_in[0];
  const float* k  = (const float*)d_in[1];
  const float* v  = (const float*)d_in[2];
  const float* wq = (const float*)d_in[3];
  const float* bq = (const float*)d_in[4];
  const float* wk = (const float*)d_in[5];
  const float* bk = (const float*)d_in[6];
  const float* wv = (const float*)d_in[7];
  const float* bv = (const float*)d_in[8];

  // Workspace: f16 Q[b][n][d], K[b][m][d], V[b][d][m]  (3 x 4 MB = 12 MB)
  _Float16* Qws = (_Float16*)d_ws;
  _Float16* Kws = Qws + (size_t)8 * 4096 * 64;
  _Float16* Vt  = Kws + (size_t)8 * 4096 * 64;

  proj_qkv_kernel<<<dim3(32, 3, 8), 256, 0, stream>>>(
      q, k, v, wq, wk, wv, bq, bk, bv, Qws, Kws, Vt);
  attn_kernel<<<dim3(32, 8), 256, 0, stream>>>(Qws, Kws, Vt, (float*)d_out);
}
